// GATv2Layer_75780402970915
// MI455X (gfx1250) — matrix operand order repaired and checked by reading, not compiled, not run on hardware
//
#include <hip/hip_runtime.h>

// ---------------------------------------------------------------------------
// GATv2 layer, fused for MI455X (gfx1250, wave32, WMMA).
//
// Math: softmax_j(where(adj, s1_i + s2_j, -9e15)) is invariant to the row
// constant s1_i, so att_ij = adj_ij * exp(s2_j - m) / sum_j adj_ij*exp(s2_j-m)
// with m = global max of s2.  The whole attention+aggregation step is then a
// single 0/1-masked GEMM:  adj[8192x8192](f16 0/1) x V[8192x80](f16), where
// V = [ w .* Wh | w | 0pad ], accumulated in f32 via v_wmma_f32_16x16x32_f16.
// Traffic is dominated by one pass over adj (256 MB -> ~11us @ 23.3 TB/s);
// the 1.3 MB V matrix stays resident in the 192 MB L2.  Each block processes
// TWO 16-row tiles so every B-tile load feeds two WMMAs (halves L2 traffic).
// adj in {0,1} -> packed f16 pair via (lo | hi<<16) * 0x3C00 (2 VALU/pair).
// ---------------------------------------------------------------------------

typedef __attribute__((ext_vector_type(16))) _Float16 v16h;
typedef __attribute__((ext_vector_type(8)))  float    v8f;

#define NN   8192
#define INF_ 128
#define OF   64
#define LRA  0.2f

// --- K1: Wh = mole @ W (f32), s2 = leakyrelu(Wh) @ b[64:128].  One block/row.
__global__ void __launch_bounds__(64) gat_wh_k(
    const float* __restrict__ mole, const float* __restrict__ W,
    const float* __restrict__ b, float* __restrict__ Wh, float* __restrict__ s2) {
  __shared__ float srow[INF_];
  __shared__ float rs[OF];
  const int row = blockIdx.x;
  const int t   = threadIdx.x;            // 64 threads = 2 waves
  srow[t]      = mole[row * INF_ + t];
  srow[t + 64] = mole[row * INF_ + 64 + t];
  __syncthreads();
  float acc = 0.f;
#pragma unroll 8
  for (int k = 0; k < INF_; ++k) acc = fmaf(srow[k], W[k * OF + t], acc);
  Wh[row * OF + t] = acc;
  const float lr = acc >= 0.f ? acc : LRA * acc;
  rs[t] = lr * b[OF + t];
  __syncthreads();
  for (int s = 32; s > 0; s >>= 1) {
    if (t < s) rs[t] += rs[t + s];
    __syncthreads();
  }
  if (t == 0) s2[row] = rs[0];
}

// --- K2: global max of s2 (one block).
__global__ void __launch_bounds__(256) gat_max_k(
    const float* __restrict__ s2, float* __restrict__ mmax) {
  __shared__ float rs[256];
  float v = -3.4e38f;
  for (int j = threadIdx.x; j < NN; j += 256) v = fmaxf(v, s2[j]);
  rs[threadIdx.x] = v;
  __syncthreads();
  for (int s = 128; s > 0; s >>= 1) {
    if (threadIdx.x < s) rs[threadIdx.x] = fmaxf(rs[threadIdx.x], rs[threadIdx.x + s]);
    __syncthreads();
  }
  if (threadIdx.x == 0) mmax[0] = rs[0];
}

// --- K3: WhxT[80][8192] f16: rows 0..63 = w_j*Wh[j][c], row 64 = w_j, 65..79 = 0.
__global__ void __launch_bounds__(256) gat_vt_k(
    const float* __restrict__ Wh, const float* __restrict__ s2,
    const float* __restrict__ mmax, _Float16* __restrict__ WhxT) {
  const int gid = blockIdx.x * 256 + threadIdx.x;   // 80*8192 total
  if (gid >= 80 * NN) return;
  const int c = gid >> 13;
  const int j = gid & (NN - 1);
  const float w = expf(s2[j] - mmax[0]);
  float v = 0.f;
  if (c < OF)       v = w * Wh[j * OF + c];
  else if (c == OF) v = w;
  WhxT[c * NN + j] = (_Float16)v;
}

// adj pair {0,1} -> packed f16 pair {0.0h,1.0h}: (lo | hi<<16) * 0x3C00.
__device__ __forceinline__ unsigned int pk01(int lo, int hi) {
  return (unsigned int)(lo | (hi << 16)) * 0x3C00u;
}

// Build one 16x32 f16 A tile (wave32 WMMA layout) from 16 adj ints per lane.
__device__ __forceinline__ v16h mkA(const int* __restrict__ p) {
  const int4 a0 = *(const int4*)(p);
  const int4 a1 = *(const int4*)(p + 4);
  const int4 a2 = *(const int4*)(p + 8);
  const int4 a3 = *(const int4*)(p + 12);
  union { v16h h; unsigned int u[8]; } A;
  A.u[0] = pk01(a0.x, a0.y);
  A.u[1] = pk01(a0.z, a0.w);
  A.u[2] = pk01(a1.x, a1.y);
  A.u[3] = pk01(a1.z, a1.w);
  A.u[4] = pk01(a2.x, a2.y);
  A.u[5] = pk01(a2.z, a2.w);
  A.u[6] = pk01(a3.x, a3.y);
  A.u[7] = pk01(a3.z, a3.w);
  return A.h;
}

// --- K4: fused masked GEMM + normalize + ELU.
// Block = 128 threads (4 wave32).  Each block owns a 32-row stripe (two 16-row
// WMMA tiles sharing every B load); waves split K (2048 columns each, 64
// iters x 32 cols).  5 N-tiles: 4 numerator + denominator (w column).
__global__ void __launch_bounds__(128) gat_attn_k(
    const int* __restrict__ adj, const _Float16* __restrict__ WhxT,
    float* __restrict__ out) {
  const int lane = threadIdx.x & 31;
  const int wave = threadIdx.x >> 5;
  const int row0 = blockIdx.x << 5;    // 32-row stripe
  const int rlo  = lane & 15;          // A: row-in-tile; B/C: col-in-tile
  const int hsel = lane >> 4;          // lane half-group

  v8f acc[2][5];
#pragma unroll
  for (int rt = 0; rt < 2; ++rt)
#pragma unroll
    for (int nt = 0; nt < 5; ++nt) acc[rt][nt] = {};

  // A layout: lanes 0-15 own K = 0-15; lanes 16-31 own K = 16-31.
  const int* arow0 = adj + (size_t)(row0 + rlo) * NN + hsel * 16;
  const int* arow1 = arow0 + (size_t)16 * NN;
  // B layout: lane n owns column n; lanes 0-15 K=0-15, lanes 16-31 K=16-31
  // -> 16 contiguous halves (32B) per lane from the transposed V matrix.
  const int kbB = hsel * 16;

  int j0 = wave * (NN / 4);
  for (int it = 0; it < NN / 4 / 32; ++it, j0 += 32) {
    const v16h A0 = mkA(arow0 + j0);
    const v16h A1 = mkA(arow1 + j0);
    __builtin_prefetch((const void*)(arow0 + j0 + 512), 0, 0);
    __builtin_prefetch((const void*)(arow1 + j0 + 512), 0, 0);

#pragma unroll
    for (int nt = 0; nt < 5; ++nt) {
      const v16h bt = *(const v16h*)(WhxT + (size_t)(nt * 16 + rlo) * NN + (j0 + kbB));
      acc[0][nt] = __builtin_amdgcn_wmma_f32_16x16x32_f16(
          false, A0, false, bt, (short)0, acc[0][nt], false, false);
      acc[1][nt] = __builtin_amdgcn_wmma_f32_16x16x32_f16(
          false, A1, false, bt, (short)0, acc[1][nt], false, false);
    }
  }

  // Cross-wave split-K reduction through LDS (C layout is identical per wave).
  __shared__ float red[3][32][80];
  if (wave > 0) {
#pragma unroll
    for (int rt = 0; rt < 2; ++rt)
#pragma unroll
      for (int nt = 0; nt < 5; ++nt)
#pragma unroll
        for (int r = 0; r < 8; ++r)
          red[wave - 1][lane][rt * 40 + nt * 8 + r] = acc[rt][nt][r];
  }
  __syncthreads();
  if (wave == 0) {
#pragma unroll
    for (int w = 0; w < 3; ++w)
#pragma unroll
      for (int rt = 0; rt < 2; ++rt)
#pragma unroll
        for (int nt = 0; nt < 5; ++nt)
#pragma unroll
          for (int r = 0; r < 8; ++r)
            acc[rt][nt][r] += red[w][lane][rt * 40 + nt * 8 + r];

#pragma unroll
    for (int rt = 0; rt < 2; ++rt) {
      // Denominator = tile 4, N=0 (lanes 0 / 16); broadcast within each half.
      float den[8];
#pragma unroll
      for (int r = 0; r < 8; ++r) den[r] = __shfl(acc[rt][4][r], hsel * 16, 32);

      // C/D layout: lanes 0-15 vgpr r -> row r; lanes 16-31 -> row r+8.
#pragma unroll
      for (int nt = 0; nt < 4; ++nt)
#pragma unroll
        for (int r = 0; r < 8; ++r) {
          const float x = acc[rt][nt][r] / den[r];
          const float y = x > 0.f ? x : expm1f(x);  // ELU (alpha = 1)
          out[(size_t)(row0 + rt * 16 + hsel * 8 + r) * OF + nt * 16 + rlo] = y;
        }
    }
  }
}

// ---------------------------------------------------------------------------
extern "C" void kernel_launch(void* const* d_in, const int* in_sizes, int n_in,
                              void* d_out, int out_size, void* d_ws, size_t ws_size,
                              hipStream_t stream) {
  (void)in_sizes; (void)n_in; (void)out_size; (void)ws_size;
  const float* mole = (const float*)d_in[0];   // [8192,128] f32
  const int*   adj  = (const int*)d_in[1];     // [8192,8192] i32 (0/1)
  const float* W    = (const float*)d_in[2];   // [128,64] f32
  const float* b    = (const float*)d_in[3];   // [128] f32
  float*       out  = (float*)d_out;           // [8192,64] f32

  char* ws = (char*)d_ws;
  float*    Wh   = (float*)ws;                                 // 2,097,152 B
  float*    s2   = (float*)(ws + 2097152);                     //    32,768 B
  float*    mmax = (float*)(ws + 2097152 + 32768);             //       256 B
  _Float16* WhxT = (_Float16*)(ws + 2097152 + 32768 + 256);    // 1,310,720 B

  gat_wh_k  <<<NN,              64,  0, stream>>>(mole, W, b, Wh, s2);
  gat_max_k <<<1,               256, 0, stream>>>(s2, mmax);
  gat_vt_k  <<<(80 * NN) / 256, 256, 0, stream>>>(Wh, s2, mmax, WhxT);
  gat_attn_k<<<NN / 32,         128, 0, stream>>>(adj, WhxT, out);
}
